// MSDGAD_44487271252061
// MI455X (gfx1250) — compile-verified, matrix operand order checked by reading, not run on hardware
//
#include <hip/hip_runtime.h>
#include <hip/hip_bf16.h>

typedef __attribute__((ext_vector_type(16))) _Float16 v16h;
typedef __attribute__((ext_vector_type(8)))  _Float16 v8h;
typedef __attribute__((ext_vector_type(8)))  float    v8f;

#define HID 64
#define TPB 256

// ---------------------------------------------------------------------------
// K-index pattern for 16-bit WMMA A/B fragments (ISA 7.12.2, 16-bit 16x32):
// lane<16: base=0, lane>=16: base=8 ; element i<8 -> k=base+i ; i>=8 -> k=base+i+8
__device__ __forceinline__ int kpat(int lane, int i) {
    int base = (lane < 16) ? 0 : 8;
    return (i < 8) ? (base + i) : (base + i + 8);
}

// ---------------------------------------------------------------------------
__global__ void deg_kernel(const int* __restrict__ src, const int* __restrict__ dst,
                           float* __restrict__ degO, float* __restrict__ degI, int E) {
    int e = blockIdx.x * TPB + threadIdx.x;
    if (e < E) {
        atomicAdd(&degO[src[e]], 1.0f);
        atomicAdd(&degI[dst[e]], 1.0f);
    }
}

__global__ void norm_kernel(const float* __restrict__ degO, const float* __restrict__ degI,
                            float* __restrict__ ns, float* __restrict__ nd, int N) {
    int i = blockIdx.x * TPB + threadIdx.x;
    if (i < N) {
        ns[i] = rsqrtf(fmaxf(degO[i], 1.0f));
        nd[i] = rsqrtf(fmaxf(degI[i], 1.0f));
    }
}

__global__ void f32_to_f16_kernel(const float* __restrict__ in, _Float16* __restrict__ out, int n) {
    int i = blockIdx.x * TPB + threadIdx.x;
    if (i < n) out[i] = (_Float16)in[i];
}

// Pack W[K x 64] (f32, row-major) into per-lane WMMA B fragments:
// out[(((s*4+t)*32 + lane)*16 + i)] = W[(s*32 + kpat(lane,i))*64 + t*16 + lane%16]
__global__ void pack_w_kernel(const float* __restrict__ W, _Float16* __restrict__ out, int K) {
    int idx = blockIdx.x * TPB + threadIdx.x;
    int total = (K / 32) * 4 * 32 * 16;
    if (idx >= total) return;
    int i = idx & 15;
    int l = (idx >> 4) & 31;
    int t = (idx >> 9) & 3;
    int s = idx >> 11;
    int k = s * 32 + kpat(l, i);
    int n = t * 16 + (l & 15);
    out[idx] = (_Float16)W[k * HID + n];
}

// ---------------------------------------------------------------------------
// Y[N,64] (f32) = X[N,K] (f16) @ Wp (packed f16 B-fragments). 1 wave = 16 rows.
template <int K>
__global__ void gemm_node_kernel(const _Float16* __restrict__ X,
                                 const _Float16* __restrict__ Wp,
                                 float* __restrict__ Y, int N) {
    const int wave = threadIdx.x >> 5;
    const int lane = threadIdx.x & 31;
    const int tile = blockIdx.x * (TPB / 32) + wave;
    int row = tile * 16 + (lane & 15);
    int rclamp = row < N ? row : N - 1;
    const _Float16* rowp = X + (size_t)rclamp * K;
    const int base = (lane < 16) ? 0 : 8;

    v8f c0 = {}, c1 = {}, c2 = {}, c3 = {};
#pragma unroll
    for (int s = 0; s < K / 32; ++s) {
        v8h lo = *(const v8h*)(rowp + s * 32 + base);
        v8h hi = *(const v8h*)(rowp + s * 32 + base + 16);
        v16h a;
#pragma unroll
        for (int i = 0; i < 8; ++i) { a[i] = lo[i]; a[i + 8] = hi[i]; }
        const _Float16* wb = Wp + ((size_t)(s * 4) * 32 + lane) * 16;
        v16h b0 = *(const v16h*)(wb);
        v16h b1 = *(const v16h*)(wb + 512);
        v16h b2 = *(const v16h*)(wb + 1024);
        v16h b3 = *(const v16h*)(wb + 1536);
        c0 = __builtin_amdgcn_wmma_f32_16x16x32_f16(false, a, false, b0, (short)0, c0, false, false);
        c1 = __builtin_amdgcn_wmma_f32_16x16x32_f16(false, a, false, b1, (short)0, c1, false, false);
        c2 = __builtin_amdgcn_wmma_f32_16x16x32_f16(false, a, false, b2, (short)0, c2, false, false);
        c3 = __builtin_amdgcn_wmma_f32_16x16x32_f16(false, a, false, b3, (short)0, c3, false, false);
    }
    // C layout: lane holds (M = v + 8*(lane/16), Ncol = t*16 + lane%16)
    const int mbase = tile * 16 + ((lane >> 4) << 3);
    const int col = lane & 15;
#pragma unroll
    for (int v = 0; v < 8; ++v) {
        int r = mbase + v;
        if (r < N) {
            float* yp = Y + (size_t)r * HID + col;
            yp[0]  = c0[v];
            yp[16] = c1[v];
            yp[32] = c2[v];
            yp[48] = c3[v];
        }
    }
}

// agg[dst[e], :] += Y[src[e], :] * norm_src[src[e]]   (16 threads x float4 per edge)
__global__ void scatter_kernel(const float* __restrict__ Y, const int* __restrict__ src,
                               const int* __restrict__ dst, const float* __restrict__ ns,
                               float* __restrict__ agg, int E) {
    long long idx = (long long)blockIdx.x * TPB + threadIdx.x;
    int e = (int)(idx >> 4);
    if (e >= E) return;
    int d = ((int)idx & 15) * 4;
    int s = src[e], dd = dst[e];
    float w = ns[s];
    const float4 v = *(const float4*)(Y + (size_t)s * HID + d);
    float* ap = agg + (size_t)dd * HID + d;
    atomicAdd(ap + 0, v.x * w);
    atomicAdd(ap + 1, v.y * w);
    atomicAdd(ap + 2, v.z * w);
    atomicAdd(ap + 3, v.w * w);
}

// out[n,d] = f16( act( agg[n,d]*norm_dst[n] + b[d] ) )
__global__ void bias_act_kernel(const float* __restrict__ agg, const float* __restrict__ nd,
                                const float* __restrict__ b, _Float16* __restrict__ out,
                                int N, int do_relu) {
    int idx = blockIdx.x * TPB + threadIdx.x;
    if (idx >= N * HID) return;
    int n = idx >> 6, d = idx & 63;
    float v = fmaf(agg[idx], nd[n], b[d]);
    if (do_relu) v = fmaxf(v, 0.0f);
    out[idx] = (_Float16)v;
}

// ---------------------------------------------------------------------------
// Scorer: per wave, 16 edges. Gather [h[src]|h[dst]] (128 f16) into LDS,
// WMMA [16x128]@[128x64], fused bias+ReLU+dot(W4)+b4, half-wave reduce, store.
__global__ void scorer_kernel(const _Float16* __restrict__ H,
                              const int* __restrict__ srcn, const int* __restrict__ dstn,
                              const _Float16* __restrict__ W3p,
                              const float* __restrict__ b3, const float* __restrict__ W4,
                              const float* __restrict__ b4,
                              float* __restrict__ out, int E) {
    __shared__ _Float16 tile[TPB / 32][16][128];
    const int wave = threadIdx.x >> 5;
    const int lane = threadIdx.x & 31;
    const int ebase = (blockIdx.x * (TPB / 32) + wave) * 16;
    const int r = lane & 15;

    int eg = ebase + r;
    if (eg >= E) eg = E - 1;
    int node = (lane < 16) ? srcn[eg] : dstn[eg];
    {
        const v8h* sp = (const v8h*)(H + (size_t)node * HID);
        v8h* dp = (v8h*)&tile[wave][r][(lane < 16) ? 0 : 64];
#pragma unroll
        for (int j = 0; j < 8; ++j) dp[j] = sp[j];
    }
    __syncthreads();

    const int base = (lane < 16) ? 0 : 8;
    v8f c[4] = {};
#pragma unroll
    for (int s = 0; s < 4; ++s) {
        const _Float16* ap = &tile[wave][r][s * 32 + base];
        v8h lo = *(const v8h*)ap;
        v8h hi = *(const v8h*)(ap + 16);
        v16h a;
#pragma unroll
        for (int i = 0; i < 8; ++i) { a[i] = lo[i]; a[i + 8] = hi[i]; }
#pragma unroll
        for (int t = 0; t < 4; ++t) {
            v16h b = *(const v16h*)(W3p + ((size_t)(s * 4 + t) * 32 + lane) * 16);
            c[t] = __builtin_amdgcn_wmma_f32_16x16x32_f16(false, a, false, b, (short)0, c[t],
                                                          false, false);
        }
    }

    const int col = lane & 15;
    float partial[8] = {0, 0, 0, 0, 0, 0, 0, 0};
#pragma unroll
    for (int t = 0; t < 4; ++t) {
        int n = t * 16 + col;
        float bn = b3[n], wn = W4[n];
#pragma unroll
        for (int v = 0; v < 8; ++v) {
            float z = fmaxf(c[t][v] + bn, 0.0f);
            partial[v] = fmaf(z, wn, partial[v]);
        }
    }
#pragma unroll
    for (int off = 1; off < 16; off <<= 1)
#pragma unroll
        for (int v = 0; v < 8; ++v) partial[v] += __shfl_xor(partial[v], off, 16);

    if (col == 0) {
        float bb = b4[0];
        int m0 = ebase + ((lane >> 4) << 3);
#pragma unroll
        for (int v = 0; v < 8; ++v) {
            int e = m0 + v;
            if (e < E) out[e] = partial[v] + bb;
        }
    }
}

// ---------------------------------------------------------------------------
extern "C" void kernel_launch(void* const* d_in, const int* in_sizes, int n_in,
                              void* d_out, int out_size, void* d_ws, size_t ws_size,
                              hipStream_t stream) {
    (void)n_in; (void)out_size; (void)ws_size;
    const float* feats   = (const float*)d_in[0];
    const float* W1      = (const float*)d_in[1];
    const float* b1      = (const float*)d_in[2];
    const float* W2      = (const float*)d_in[3];
    const float* b2      = (const float*)d_in[4];
    const float* W3      = (const float*)d_in[5];
    const float* b3      = (const float*)d_in[6];
    const float* W4      = (const float*)d_in[7];
    const float* b4      = (const float*)d_in[8];
    const int* src_seq   = (const int*)d_in[9];
    const int* dst_seq   = (const int*)d_in[10];
    const int* src_next  = (const int*)d_in[11];
    const int* dst_next  = (const int*)d_in[12];

    const int E = in_sizes[11];
    const int T = in_sizes[9] / E;
    const int N = in_sizes[0] / (T * 128);
    // only the last snapshot feeds the output (h_seq[-1])
    const int* src2 = src_seq + (size_t)(T - 1) * E;
    const int* dst2 = dst_seq + (size_t)(T - 1) * E;
    const float* feats2 = feats + (size_t)(T - 1) * N * 128;

    // carve workspace (all 256B aligned)
    char* wsp = (char*)d_ws;
    auto carve = [&](size_t bytes) {
        char* p = wsp;
        wsp += (bytes + 255) & ~(size_t)255;
        return p;
    };
    float*    degO    = (float*)carve((size_t)N * 4);
    float*    degI    = (float*)carve((size_t)N * 4);
    float*    nrmS    = (float*)carve((size_t)N * 4);
    float*    nrmD    = (float*)carve((size_t)N * 4);
    _Float16* feats16 = (_Float16*)carve((size_t)N * 128 * 2);
    float*    Y       = (float*)carve((size_t)N * HID * 4);
    float*    agg     = (float*)carve((size_t)N * HID * 4);
    _Float16* h1_16   = (_Float16*)carve((size_t)N * HID * 2);
    _Float16* h2_16   = (_Float16*)carve((size_t)N * HID * 2);
    _Float16* W1p     = (_Float16*)carve(8192 * 2);
    _Float16* W2p     = (_Float16*)carve(4096 * 2);
    _Float16* W3p     = (_Float16*)carve(8192 * 2);

    const int gE   = (E + TPB - 1) / TPB;
    const int gN   = (N + TPB - 1) / TPB;
    const int gNF  = (N * 128 + TPB - 1) / TPB;
    const int gNH  = (N * HID + TPB - 1) / TPB;
    const int gSc  = (int)(((long long)E * 16 + TPB - 1) / TPB);
    const int gGem = (N + 127) / 128;          // 8 waves x 16 rows per block
    const int gScore = (E + 127) / 128;        // 8 waves x 16 edges per block

    // ---- degrees & norms
    hipMemsetAsync(degO, 0, (size_t)N * 4, stream);
    hipMemsetAsync(degI, 0, (size_t)N * 4, stream);
    deg_kernel<<<gE, TPB, 0, stream>>>(src2, dst2, degO, degI, E);
    norm_kernel<<<gN, TPB, 0, stream>>>(degO, degI, nrmS, nrmD, N);

    // ---- precision convert + weight fragment packing
    f32_to_f16_kernel<<<gNF, TPB, 0, stream>>>(feats2, feats16, N * 128);
    pack_w_kernel<<<(8192 + TPB - 1) / TPB, TPB, 0, stream>>>(W1, W1p, 128);
    pack_w_kernel<<<(4096 + TPB - 1) / TPB, TPB, 0, stream>>>(W2, W2p, 64);
    pack_w_kernel<<<(8192 + TPB - 1) / TPB, TPB, 0, stream>>>(W3, W3p, 128);

    // ---- GCN layer 1:  y = X@W1 ; agg = scatter(y*ns) ; h1 = relu(agg*nd + b1)
    gemm_node_kernel<128><<<gGem, TPB, 0, stream>>>(feats16, W1p, Y, N);
    hipMemsetAsync(agg, 0, (size_t)N * HID * 4, stream);
    scatter_kernel<<<gSc, TPB, 0, stream>>>(Y, src2, dst2, nrmS, agg, E);
    bias_act_kernel<<<gNH, TPB, 0, stream>>>(agg, nrmD, b1, h1_16, N, 1);

    // ---- GCN layer 2:  y = h1@W2 ; agg = scatter(y*ns) ; h = agg*nd + b2
    gemm_node_kernel<64><<<gGem, TPB, 0, stream>>>(h1_16, W2p, Y, N);
    hipMemsetAsync(agg, 0, (size_t)N * HID * 4, stream);
    scatter_kernel<<<gSc, TPB, 0, stream>>>(Y, src2, dst2, nrmS, agg, E);
    bias_act_kernel<<<gNH, TPB, 0, stream>>>(agg, nrmD, b2, h2_16, N, 0);

    // ---- edge scorer: relu([h[src]|h[dst]]@W3 + b3) @ W4 + b4
    scorer_kernel<<<gScore, TPB, 0, stream>>>(h2_16, src_next, dst_next, W3p, b3, W4, b4,
                                              (float*)d_out, E);
}
